// VectorQuantizer2_77773267796211
// MI455X (gfx1250) — matrix operand order repaired and checked by reading, not compiled, not run on hardware
//
#include <hip/hip_runtime.h>
#include <hip/hip_bf16.h>

// ---------------------------------------------------------------------------
// VAR multi-scale vector quantizer for MI455X (gfx1250, wave32, WMMA)
// B=64, C=32, H=W=16, V=4096, scales {1,2,4,6,8,10,13,16}
// Distance GEMM: split-bf16 (hi+lo) on v_wmma_f32_16x16x32_bf16,
//   r.e ~= r_hi.e_hi + r_hi.e_lo + r_lo.e_hi   (near-f32 accuracy)
// ---------------------------------------------------------------------------

typedef __attribute__((ext_vector_type(16))) __bf16 v16bf;
typedef __attribute__((ext_vector_type(8)))  __bf16 v8bf;
typedef __attribute__((ext_vector_type(8)))  float  v8f;

#define HWD   16
#define C_DIM 32
#define B_DIM 64
#define V_DIM 4096
#define SN    8

__device__ __forceinline__ unsigned short f2bf(float x) {
    unsigned u = __float_as_uint(x);
    u += 0x7FFFu + ((u >> 16) & 1u);      // round-to-nearest-even
    return (unsigned short)(u >> 16);
}
__device__ __forceinline__ float bf2f(unsigned short h) {
    return __uint_as_float((unsigned)h << 16);
}
// split x into bf16 hi + bf16 lo (x ~= hi + lo)
__device__ __forceinline__ void f2bf2(float x, unsigned short* hi, unsigned short* lo) {
    unsigned short h = f2bf(x);
    *hi = h;
    *lo = f2bf(x - bf2f(h));
}

// --- Kernel 1: codebook -> split bf16 + per-code squared norm --------------
__global__ void vq_prep_embed(const float* __restrict__ embed,
                              unsigned short* __restrict__ ehi,
                              unsigned short* __restrict__ elo,
                              float* __restrict__ esq) {
    int v = blockIdx.x * blockDim.x + threadIdx.x;
    if (v >= V_DIM) return;
    float s = 0.f;
    #pragma unroll
    for (int c = 0; c < C_DIM; ++c) {
        float e = embed[v * C_DIM + c];
        s += e * e;
        unsigned short h, l;
        f2bf2(e, &h, &l);
        ehi[v * C_DIM + c] = h;
        elo[v * C_DIM + c] = l;
    }
    esq[v] = s;
}

// --- Kernel 2: residual + adaptive avg-pool -> split bf16 rows [N, 32] -----
__global__ void vq_pool_rest(const float* __restrict__ f,
                             const float* __restrict__ f_hat,
                             unsigned short* __restrict__ rhi,
                             unsigned short* __restrict__ rlo,
                             int pn) {
    int t = blockIdx.x * blockDim.x + threadIdx.x;
    int total = B_DIM * pn * pn * C_DIM;
    if (t >= total) return;
    int c   = t & (C_DIM - 1);
    int row = t >> 5;                     // (b*pn + oy)*pn + ox
    int ox  = row % pn;
    int tmp = row / pn;
    int oy  = tmp % pn;
    int b   = tmp / pn;
    int sy = (oy * HWD) / pn, ey = ((oy + 1) * HWD + pn - 1) / pn;
    int sx = (ox * HWD) / pn, ex = ((ox + 1) * HWD + pn - 1) / pn;
    const float* fp = f     + (size_t)(b * C_DIM + c) * (HWD * HWD);
    const float* hp = f_hat + (size_t)(b * C_DIM + c) * (HWD * HWD);
    float s = 0.f;
    for (int yy = sy; yy < ey; ++yy)
        for (int xx = sx; xx < ex; ++xx)
            s += fp[yy * HWD + xx] - hp[yy * HWD + xx];
    s *= 1.0f / (float)((ey - sy) * (ex - sx));
    unsigned short h, l;
    f2bf2(s, &h, &l);
    rhi[(size_t)row * C_DIM + c] = h;
    rlo[(size_t)row * C_DIM + c] = l;
}

// A-tile loader: lane holds row m, K in [16g, 16g+16)
__device__ __forceinline__ v16bf load_a_tile(const unsigned short* base,
                                             int row, int g) {
    const unsigned short* ar = base + (size_t)row * C_DIM + 16 * g;
    v8bf lo = *reinterpret_cast<const v8bf*>(ar);
    v8bf hi = *reinterpret_cast<const v8bf*>(ar + 8);
    return __builtin_shufflevector(lo, hi,
        0, 1, 2, 3, 4, 5, 6, 7, 8, 9, 10, 11, 12, 13, 14, 15);
}

// --- Kernel 3: WMMA codebook argmin ----------------------------------------
// One wave handles a 16-row tile; loops 256 code tiles of 16 codes.
// score = ||e||^2 - 2 * rest . e   (row-constant ||rest||^2 dropped)
__global__ void vq_argmin_wmma(const unsigned short* __restrict__ rhi,
                               const unsigned short* __restrict__ rlo,
                               const unsigned short* __restrict__ ehi,
                               const unsigned short* __restrict__ elo,
                               const float* __restrict__ esq,
                               int* __restrict__ idx_out, int N) {
    int gtid = blockIdx.x * blockDim.x + threadIdx.x;
    int wave = gtid >> 5;                 // tile id
    int lane = threadIdx.x & 31;
    int m = lane & 15;                    // A row / B column within tile
    int g = lane >> 4;                    // half-wave selector
    if (wave * 16 >= N) return;           // wave-uniform

    v16bf a_hi = load_a_tile(rhi, wave * 16 + m, g);
    v16bf a_lo = load_a_tile(rlo, wave * 16 + m, g);

    float best[8];
    int   bidx[8];
    #pragma unroll
    for (int r = 0; r < 8; ++r) { best[r] = 3.4e38f; bidx[r] = 0; }

    #pragma unroll 2
    for (int t = 0; t < V_DIM / 16; ++t) {
        int n = t * 16 + m;               // code index for this lane's column
        // B tile: code n, K in [16g, 16g+16) -- contiguous 32 bytes
        size_t boff = (size_t)n * C_DIM + 16 * g;
        v16bf b_hi = *reinterpret_cast<const v16bf*>(ehi + boff);
        v16bf b_lo = *reinterpret_cast<const v16bf*>(elo + boff);
        v8f c = {};
        c = __builtin_amdgcn_wmma_f32_16x16x32_bf16(
            false, a_hi, false, b_hi, (short)0, c, false, false);
        c = __builtin_amdgcn_wmma_f32_16x16x32_bf16(
            false, a_hi, false, b_lo, (short)0, c, false, false);
        c = __builtin_amdgcn_wmma_f32_16x16x32_bf16(
            false, a_lo, false, b_hi, (short)0, c, false, false);
        float es = esq[n];
        #pragma unroll
        for (int r = 0; r < 8; ++r) {
            float s = __builtin_fmaf(-2.0f, c[r], es);
            if (s < best[r]) { best[r] = s; bidx[r] = n; }
        }
    }
    // reduce across the 16 column-lanes (xor 1..8 stays within each half)
    #pragma unroll
    for (int off = 1; off < 16; off <<= 1) {
        #pragma unroll
        for (int r = 0; r < 8; ++r) {
            float os = __shfl_xor(best[r], off, 32);
            int   oi = __shfl_xor(bidx[r], off, 32);
            if (os < best[r] || (os == best[r] && oi < bidx[r])) {
                best[r] = os; bidx[r] = oi;
            }
        }
    }
    if (m == 0) {
        #pragma unroll
        for (int r = 0; r < 8; ++r) {
            int row = wave * 16 + r + 8 * g;   // D layout: M = r + 8*(lane/16)
            if (row < N) idx_out[row] = bidx[r];
        }
    }
}

// --- Kernel 4: gather + bilinear upsample + phi(conv3x3) + accumulate ------
__global__ void vq_phi_accum(const float* __restrict__ f,
                             const float* __restrict__ embed,
                             const int* __restrict__ idxbuf,
                             const float* __restrict__ w,   // [32][32][3][3]
                             const float* __restrict__ bias,// [32]
                             float* __restrict__ f_hat,
                             float* __restrict__ loss_acc,
                             int pn, float loss_scale) {
    __shared__ float sh_h[C_DIM * HWD * HWD];   // 32 KB: upsampled h
    __shared__ float sh_red[8];
    int b   = blockIdx.x;
    int tid = threadIdx.x;                      // 256 threads = one pixel each
    int y = tid >> 4, x = tid & 15;

    // bilinear source coords, align_corners=False (half-pixel centers)
    float scl = (float)pn / (float)HWD;
    float fy = (y + 0.5f) * scl - 0.5f;
    float fx = (x + 0.5f) * scl - 0.5f;
    int y0i = (int)floorf(fy); float wy = fy - (float)y0i;
    int x0i = (int)floorf(fx); float wx = fx - (float)x0i;
    int y1i = min(max(y0i + 1, 0), pn - 1);
    int x1i = min(max(x0i + 1, 0), pn - 1);
    y0i = min(max(y0i, 0), pn - 1);
    x0i = min(max(x0i, 0), pn - 1);
    int base = b * pn * pn;
    int i00 = idxbuf[base + y0i * pn + x0i];
    int i01 = idxbuf[base + y0i * pn + x1i];
    int i10 = idxbuf[base + y1i * pn + x0i];
    int i11 = idxbuf[base + y1i * pn + x1i];
    float w00 = (1.f - wy) * (1.f - wx), w01 = (1.f - wy) * wx;
    float w10 = wy * (1.f - wx),         w11 = wy * wx;

    #pragma unroll
    for (int c = 0; c < C_DIM; ++c) {
        float h = w00 * embed[i00 * C_DIM + c] + w01 * embed[i01 * C_DIM + c]
                + w10 * embed[i10 * C_DIM + c] + w11 * embed[i11 * C_DIM + c];
        sh_h[c * 256 + tid] = h;
    }
    __syncthreads();

    float lsum = 0.f;
    for (int co = 0; co < C_DIM; ++co) {
        float acc = bias[co];
        const float* wc = w + co * C_DIM * 9;   // uniform -> scalar loads
        for (int ci = 0; ci < C_DIM; ++ci) {
            const float* hp = &sh_h[ci * 256];
            const float* wk = wc + ci * 9;
            #pragma unroll
            for (int ky = 0; ky < 3; ++ky) {
                int yy = y + ky - 1;
                if (yy < 0 || yy > 15) continue;
                #pragma unroll
                for (int kx = 0; kx < 3; ++kx) {
                    int xx = x + kx - 1;
                    if (xx < 0 || xx > 15) continue;
                    acc = __builtin_fmaf(wk[ky * 3 + kx], hp[yy * HWD + xx], acc);
                }
            }
        }
        float hval = sh_h[co * 256 + tid];
        float outv = 0.5f * hval + 0.5f * acc;          // QUANT_RESI = 0.5
        size_t gi = (size_t)(b * C_DIM + co) * (HWD * HWD) + tid;
        float fh = f_hat[gi] + outv;
        f_hat[gi] = fh;
        float d = fh - f[gi];
        lsum = __builtin_fmaf(d, d, lsum);
    }
    // block loss reduction
    #pragma unroll
    for (int off = 16; off > 0; off >>= 1)
        lsum += __shfl_down(lsum, off, 32);
    if ((tid & 31) == 0) sh_red[tid >> 5] = lsum;
    __syncthreads();
    if (tid == 0) {
        float s = 0.f;
        #pragma unroll
        for (int i = 0; i < 8; ++i) s += sh_red[i];
        atomicAdd(loss_acc, s * loss_scale);
    }
}

// --- Kernel 5: write scalar loss into d_out tail ---------------------------
__global__ void vq_finalize(const float* __restrict__ loss_acc,
                            float* __restrict__ out_loss) {
    if (threadIdx.x == 0) out_loss[0] = loss_acc[0];
}

// ---------------------------------------------------------------------------
extern "C" void kernel_launch(void* const* d_in, const int* in_sizes, int n_in,
                              void* d_out, int out_size, void* d_ws, size_t ws_size,
                              hipStream_t stream) {
    const float* f      = (const float*)d_in[0];   // [64,32,16,16]
    const float* embed  = (const float*)d_in[1];   // [4096,32]
    const float* phi_w  = (const float*)d_in[2];   // [4,32,32,3,3]
    const float* phi_b  = (const float*)d_in[3];   // [4,32]
    float* out   = (float*)d_out;
    float* f_hat = out;                            // first 524288 floats

    const size_t EB = (size_t)V_DIM * C_DIM * 2;   // 256 KB per embed plane
    const size_t RB = (size_t)16384 * C_DIM * 2;   // 1 MB per rest plane
    char* ws = (char*)d_ws;
    float*          loss_acc = (float*)ws;                          // pad 256 B
    unsigned short* ehi      = (unsigned short*)(ws + 256);
    unsigned short* elo      = (unsigned short*)(ws + 256 + EB);
    float*          esq      = (float*)         (ws + 256 + 2 * EB);
    unsigned short* rhi      = (unsigned short*)(ws + 256 + 2 * EB + V_DIM * 4);
    unsigned short* rlo      = (unsigned short*)(ws + 256 + 2 * EB + V_DIM * 4 + RB);
    int*            idxb     = (int*)           (ws + 256 + 2 * EB + V_DIM * 4 + 2 * RB);

    const size_t nf = (size_t)B_DIM * C_DIM * HWD * HWD;           // 524288
    hipMemsetAsync(d_out, 0, (size_t)out_size * sizeof(float), stream); // f_hat = 0
    hipMemsetAsync(loss_acc, 0, sizeof(float), stream);

    vq_prep_embed<<<V_DIM / 256, 256, 0, stream>>>(embed, ehi, elo, esq);

    const int pns[SN] = {1, 2, 4, 6, 8, 10, 13, 16};
    const float loss_scale = 1.25f / ((float)nf * (float)SN);
    for (int si = 0; si < SN; ++si) {
        int pn = pns[si];
        int N  = B_DIM * pn * pn;          // rows (always multiple of 64)
        int total = N * C_DIM;
        vq_pool_rest<<<(total + 255) / 256, 256, 0, stream>>>(f, f_hat, rhi, rlo, pn);
        // one wave per 16-row tile; 4 waves per block => N/64 blocks
        vq_argmin_wmma<<<N / 64, 128, 0, stream>>>(rhi, rlo, ehi, elo, esq, idxb, N);
        int k = si >> 1;                   // PhiPartiallyShared tick index
        vq_phi_accum<<<B_DIM, 256, 0, stream>>>(
            f, embed, idxb,
            phi_w + (size_t)k * C_DIM * C_DIM * 9,
            phi_b + (size_t)k * C_DIM,
            f_hat, loss_acc, pn, loss_scale);
    }
    vq_finalize<<<1, 1, 0, stream>>>(loss_acc, out + nf);
}
